// CustomAttention_37306085933142
// MI455X (gfx1250) — matrix operand order repaired and checked by reading, not compiled, not run on hardware
//
#include <hip/hip_runtime.h>
#include <hip/hip_bf16.h>

typedef unsigned int u32;
typedef unsigned short u16;

typedef __attribute__((ext_vector_type(16))) __bf16 v16bf;
typedef __attribute__((ext_vector_type(8)))  float  v8f;
typedef __attribute__((ext_vector_type(4)))  u32    v4u;

union V16 {
  v16bf v;
  u32   u[8];
  v4u   q[2];
};

#define B_  8
#define T_  1024
#define D_  512
#define H_  8
#define DK_ 64
#define P_  2047

#define WMMA_BF16(A, Bm, C) \
  __builtin_amdgcn_wmma_f32_16x16x32_bf16(false, (A), false, (Bm), (short)0, (C), false, false)

__device__ __forceinline__ u16 f2bf(float f) {
  u32 u = __float_as_uint(f);
  u += 0x7FFFu + ((u >> 16) & 1u);   // round-to-nearest-even
  return (u16)(u >> 16);
}
__device__ __forceinline__ u32 pack2bf(float lo, float hi) {
  return (u32)f2bf(lo) | ((u32)f2bf(hi) << 16);
}
__device__ __forceinline__ float bperm(int srcLane, float v) {
  return __int_as_float(__builtin_amdgcn_ds_bpermute(srcLane << 2, __float_as_int(v)));
}
__device__ __forceinline__ float red_max16(float v, int lane) {
#pragma unroll
  for (int m = 1; m < 16; m <<= 1) v = fmaxf(v, bperm(lane ^ m, v));
  return v;
}
__device__ __forceinline__ float red_sum16(float v, int lane) {
#pragma unroll
  for (int m = 1; m < 16; m <<= 1) v += bperm(lane ^ m, v);
  return v;
}

// ---------------------------------------------------------------------------
// f32 -> bf16 elementwise convert
// ---------------------------------------------------------------------------
__global__ void cvt_bf16_kernel(const float* __restrict__ src,
                                u16* __restrict__ dst, int n) {
  int i = blockIdx.x * 256 + threadIdx.x;
  if (i < n) dst[i] = f2bf(src[i]);
}

// ---------------------------------------------------------------------------
// WMMA projection GEMM:  Y[i,j] = sum_d X[i,d] * W[j,d]  (+bias[j])
// One wave computes a 32x32 block (2x2 grid of 16x16 WMMA tiles) so each
// k-step does 4 WMMAs on 4 operand loads -> ~16 FLOP/B from L2.
// mode 0: Q  -> write q_u,q_v bf16 at (b,h,t,dk) with +pos_bias_u/v
// mode 1: K  -> write k   bf16 at (b,h,t,dk)
// mode 2: V  -> write v^T bf16 at (b,h,dk,t)
// mode 3: P  -> write p   bf16 at (h,pos,dk)   (M padded to 2048)
// mode 4: O  -> write f32 to out (+bo)
// ---------------------------------------------------------------------------
__global__ __launch_bounds__(128) void proj_kernel(
    const u16* __restrict__ X, const u16* __restrict__ W,
    const float* __restrict__ bias, int M, int Mreal, int Kdim, int mode,
    const float* __restrict__ pbu, const float* __restrict__ pbv,
    u16* __restrict__ outA, u16* __restrict__ outB, float* __restrict__ outF) {
  const int lane = threadIdx.x & 31;
  const int wave = threadIdx.x >> 5;
  const int half = lane >> 4;
  const int nn   = lane & 15;
  const int ntx  = D_ >> 5;                 // 16 col blocks of 32
  const int tile = blockIdx.x * 4 + wave;
  const int tm = tile / ntx, tn = tile % ntx;
  if (tm * 32 >= M) return;

  int ar0 = tm * 32 + nn;       if (ar0 > Mreal - 1) ar0 = Mreal - 1;
  int ar1 = tm * 32 + 16 + nn;  if (ar1 > Mreal - 1) ar1 = Mreal - 1;
  const v4u* xr0 = (const v4u*)(X + (size_t)ar0 * Kdim);
  const v4u* xr1 = (const v4u*)(X + (size_t)ar1 * Kdim);
  const v4u* wr0 = (const v4u*)(W + (size_t)(tn * 32 + nn) * Kdim);
  const v4u* wr1 = (const v4u*)(W + (size_t)(tn * 32 + 16 + nn) * Kdim);

  v8f cc[2][2] = {{{}, {}}, {{}, {}}};
#pragma unroll 2
  for (int kc = 0; kc < Kdim; kc += 32) {
    V16 a0, a1, b0, b1;
    // A (16x32 bf16): lane m = row, K = half*16 + e (contiguous along d)
    a0.q[0] = xr0[(kc >> 3) + 2 * half];  a0.q[1] = xr0[(kc >> 3) + 2 * half + 1];
    a1.q[0] = xr1[(kc >> 3) + 2 * half];  a1.q[1] = xr1[(kc >> 3) + 2 * half + 1];
    // B (32x16 bf16): lane n = col, K = half*16 + e (contiguous along d)
    b0.q[0] = wr0[(kc >> 3) + 2 * half];  b0.q[1] = wr0[(kc >> 3) + 2 * half + 1];
    b1.q[0] = wr1[(kc >> 3) + 2 * half];  b1.q[1] = wr1[(kc >> 3) + 2 * half + 1];
    cc[0][0] = WMMA_BF16(a0.v, b0.v, cc[0][0]);
    cc[0][1] = WMMA_BF16(a0.v, b1.v, cc[0][1]);
    cc[1][0] = WMMA_BF16(a1.v, b0.v, cc[1][0]);
    cc[1][1] = WMMA_BF16(a1.v, b1.v, cc[1][1]);
  }

#pragma unroll
  for (int mi = 0; mi < 2; ++mi) {
#pragma unroll
    for (int ni = 0; ni < 2; ++ni) {
      const int j = tn * 32 + ni * 16 + nn;
      const int h = j >> 6, dk = j & 63;
#pragma unroll
      for (int r = 0; r < 8; ++r) {
        const int i = tm * 32 + mi * 16 + r + half * 8;  // C/D layout: M = r + half*8
        if (i >= Mreal) continue;
        float val = cc[mi][ni][r];
        if (bias) val += bias[j];
        if (mode == 0) {
          const int b0i = i >> 10, t = i & (T_ - 1);
          const size_t o = ((size_t)(b0i * H_ + h) * T_ + t) * DK_ + dk;
          outA[o] = f2bf(val + pbu[h * DK_ + dk]);
          outB[o] = f2bf(val + pbv[h * DK_ + dk]);
        } else if (mode == 1) {
          const int b0i = i >> 10, t = i & (T_ - 1);
          outA[((size_t)(b0i * H_ + h) * T_ + t) * DK_ + dk] = f2bf(val);
        } else if (mode == 2) {
          const int b0i = i >> 10, t = i & (T_ - 1);
          outA[((size_t)(b0i * H_ + h) * DK_ + dk) * T_ + t] = f2bf(val);
        } else if (mode == 3) {
          outA[((size_t)h * P_ + i) * DK_ + dk] = f2bf(val);
        } else {
          outF[(size_t)i * D_ + j] = val;
        }
      }
    }
  }
}

// ---------------------------------------------------------------------------
// Fused relative-position flash attention.
// One wave owns 16 query rows of one (b,h). Keys processed 32 at a time.
// scores[t,s] = (q_u[t].k[s] + q_v[t].p[s-t+T-1]) / 8, mask -> -1e4 / 0.
// rel_shift realized as a per-lane ds_bpermute gather inside the WMMA C layout.
// ---------------------------------------------------------------------------
__global__ __launch_bounds__(128) void attn_kernel(
    const u16* __restrict__ qu, const u16* __restrict__ qv,
    const u16* __restrict__ kb, const u16* __restrict__ vT,
    const u16* __restrict__ pb, const int* __restrict__ mask,
    u16* __restrict__ ctx) {
  const int lane = threadIdx.x & 31;
  const int wave = threadIdx.x >> 5;
  const int half = lane >> 4;
  const int nn   = lane & 15;
  const int wid  = blockIdx.x * 4 + wave;          // 0..4095
  const int ttile = wid & 63;
  const int h     = (wid >> 6) & 7;
  const int b     = wid >> 9;
  const int t0    = ttile * 16;
  const size_t bh = (size_t)b * H_ + h;

  __shared__ float sc_all[4][16][34];              // per-wave transpose scratch
  float (*sc)[34] = sc_all[wave];

  // Persistent A operands: q_u, q_v tiles (16x64), two K=32 chunks each.
  const v4u* qur = (const v4u*)(qu + (bh * T_ + t0 + nn) * DK_);
  const v4u* qvr = (const v4u*)(qv + (bh * T_ + t0 + nn) * DK_);
  V16 au[2], av[2];
#pragma unroll
  for (int kc = 0; kc < 2; ++kc) {
    au[kc].q[0] = qur[kc * 4 + 2 * half];
    au[kc].q[1] = qur[kc * 4 + 2 * half + 1];
    av[kc].q[0] = qvr[kc * 4 + 2 * half];
    av[kc].q[1] = qvr[kc * 4 + 2 * half + 1];
  }

  v8f o0 = {}, o1 = {}, o2 = {}, o3 = {};
  float mrow[8], lrow[8];
#pragma unroll
  for (int r = 0; r < 8; ++r) { mrow[r] = -3.0e38f; lrow[r] = 0.f; }

  const u16* kbase = kb + bh * (size_t)T_ * DK_;
  const u16* vbase = vT + bh * (size_t)DK_ * T_;
  const u16* pbase = pb + (size_t)h * P_ * DK_;

  for (int s0 = 0; s0 < T_; s0 += 32) {
    const int j0 = s0 - t0 + (T_ - 16);            // window base, >= 0

    // ---- prefetch next step's key / value / pos streams (global_prefetch_b8)
    {
      int sn = s0 + 32; if (sn > T_ - 32) sn = 0;
      __builtin_prefetch(kbase + (size_t)(sn + lane) * DK_, 0, 3);
      __builtin_prefetch(vbase + (size_t)(lane * 2) * T_ + sn, 0, 3);
      int jn = j0 + 32 + lane; if (jn > P_ - 1) jn = P_ - 1;
      __builtin_prefetch(pbase + (size_t)jn * DK_, 0, 3);
    }

    // ---- three p-window products: cp_w[m,n] = q_v[t0+m] . p[j0+16w+n]
    v8f cp0 = {}, cp1 = {}, cp2 = {};
#pragma unroll
    for (int kc = 0; kc < 2; ++kc) {
      V16 bp0, bp1, bp2;
      int j2 = j0 + 32 + nn;
      if (j2 > P_ - 1) j2 = P_ - 1;                // clamp single OOB row
      const v4u* pr0 = (const v4u*)(pbase + (size_t)(j0 + nn) * DK_);
      const v4u* pr1 = (const v4u*)(pbase + (size_t)(j0 + 16 + nn) * DK_);
      const v4u* pr2 = (const v4u*)(pbase + (size_t)j2 * DK_);
      const int qi = kc * 4 + 2 * half;
      bp0.q[0] = pr0[qi]; bp0.q[1] = pr0[qi + 1];
      bp1.q[0] = pr1[qi]; bp1.q[1] = pr1[qi + 1];
      bp2.q[0] = pr2[qi]; bp2.q[1] = pr2[qi + 1];
      cp0 = WMMA_BF16(av[kc].v, bp0.v, cp0);
      cp1 = WMMA_BF16(av[kc].v, bp1.v, cp1);
      cp2 = WMMA_BF16(av[kc].v, bp2.v, cp2);
    }

    // ---- two 16x16 score subtiles: AC via WMMA, BD via bpermute rel-shift
    float ps[2][8], pz[2][8];
#pragma unroll
    for (int sub = 0; sub < 2; ++sub) {
      const int lsb = sub * 16;
      v8f cac = {};
#pragma unroll
      for (int kc = 0; kc < 2; ++kc) {
        V16 bk;
        const v4u* kr = (const v4u*)(kbase + (size_t)(s0 + lsb + nn) * DK_);
        bk.q[0] = kr[kc * 4 + 2 * half];
        bk.q[1] = kr[kc * 4 + 2 * half + 1];
        cac = WMMA_BF16(au[kc].v, bk.v, cac);
      }
      const v8f clo = sub ? cp1 : cp0;
      const v8f chi = sub ? cp2 : cp1;
#pragma unroll
      for (int r = 0; r < 8; ++r) {
        const int dt = r + half * 8;
        const int W  = lsb + nn - dt + 15;         // window column, in [lsb, lsb+30]
        const int src = (lane & 16) + (W & 15);    // same-row gather inside C layout
        const float blo = bperm(src, clo[r]);
        const float bhi = bperm(src, chi[r]);
        const int selhi = sub ? (W & 32) : (W & 16);
        const float bd = selhi ? bhi : blo;
        const float sval = (cac[r] + bd) * 0.125f; // 1/sqrt(64)
        const int mk = mask[((size_t)b * T_ + t0 + dt) * T_ + s0 + lsb + nn];
        ps[sub][r] = mk ? -10000.f : sval;
        pz[sub][r] = mk ? 0.f : 1.f;
      }
    }

    // ---- online softmax update + write P tile (C layout) to LDS
#pragma unroll
    for (int r = 0; r < 8; ++r) {
      const float mx   = red_max16(fmaxf(ps[0][r], ps[1][r]), lane);
      const float mnew = fmaxf(mrow[r], mx);
      const float alpha = __expf(mrow[r] - mnew);
      const float p0 = __expf(ps[0][r] - mnew) * pz[0][r];
      const float p1 = __expf(ps[1][r] - mnew) * pz[1][r];
      const float rs = red_sum16(p0 + p1, lane);
      lrow[r] = lrow[r] * alpha + rs;
      mrow[r] = mnew;
      sc[r + half * 8][nn]      = p0;
      sc[r + half * 8][16 + nn] = p1;
      o0[r] *= alpha; o1[r] *= alpha; o2[r] *= alpha; o3[r] *= alpha;
    }

    // ---- read back in A layout (16x32, K = key index), pack to bf16
    V16 pa;
#pragma unroll
    for (int r8 = 0; r8 < 8; ++r8) {
      const int K = 16 * half + 2 * r8;
      pa.u[r8] = pack2bf(sc[nn][K], sc[nn][K + 1]);
    }

    // ---- O += P @ V : 4 d-chunks, one 16x16x32 WMMA each (B from V^T rows)
    {
      const int vq = (s0 >> 3) + 2 * half;
      V16 b0, b1, b2, b3;
      const v4u* vr0 = (const v4u*)(vbase + (size_t)(0  + nn) * T_);
      const v4u* vr1 = (const v4u*)(vbase + (size_t)(16 + nn) * T_);
      const v4u* vr2 = (const v4u*)(vbase + (size_t)(32 + nn) * T_);
      const v4u* vr3 = (const v4u*)(vbase + (size_t)(48 + nn) * T_);
      b0.q[0] = vr0[vq]; b0.q[1] = vr0[vq + 1];
      b1.q[0] = vr1[vq]; b1.q[1] = vr1[vq + 1];
      b2.q[0] = vr2[vq]; b2.q[1] = vr2[vq + 1];
      b3.q[0] = vr3[vq]; b3.q[1] = vr3[vq + 1];
      o0 = WMMA_BF16(pa.v, b0.v, o0);
      o1 = WMMA_BF16(pa.v, b1.v, o1);
      o2 = WMMA_BF16(pa.v, b2.v, o2);
      o3 = WMMA_BF16(pa.v, b3.v, o3);
    }
  }

  // ---- normalize and emit context bf16 at (b, t, h, dk)
#pragma unroll
  for (int r = 0; r < 8; ++r) {
    const int t = t0 + r + half * 8;
    const float inv = 1.0f / fmaxf(lrow[r], 1e-20f);
    const size_t base = ((size_t)(b * T_ + t) * H_ + h) * DK_ + nn;
    ctx[base + 0]  = f2bf(o0[r] * inv);
    ctx[base + 16] = f2bf(o1[r] * inv);
    ctx[base + 32] = f2bf(o2[r] * inv);
    ctx[base + 48] = f2bf(o3[r] * inv);
  }
}

// ---------------------------------------------------------------------------
extern "C" void kernel_launch(void* const* d_in, const int* in_sizes, int n_in,
                              void* d_out, int out_size, void* d_ws, size_t ws_size,
                              hipStream_t stream) {
  (void)in_sizes; (void)n_in; (void)out_size; (void)ws_size;
  const float* query = (const float*)d_in[0];
  const float* key   = (const float*)d_in[1];
  const float* value = (const float*)d_in[2];
  const int*   mask  = (const int*)d_in[3];
  const float* pos   = (const float*)d_in[4];
  const float* Wq = (const float*)d_in[5];
  const float* bq = (const float*)d_in[6];
  const float* Wk = (const float*)d_in[7];
  const float* bk = (const float*)d_in[8];
  const float* Wv = (const float*)d_in[9];
  const float* bv = (const float*)d_in[10];
  const float* Wp = (const float*)d_in[11];
  const float* Wo = (const float*)d_in[12];
  const float* bo = (const float*)d_in[13];
  const float* pbu = (const float*)d_in[14];
  const float* pbv = (const float*)d_in[15];
  float* out = (float*)d_out;

  char* w = (char*)d_ws;
  auto alloc = [&](size_t bytes) -> void* {
    void* p = (void*)w;
    w += (bytes + 255) & ~(size_t)255;
    return p;
  };
  const size_t nX = (size_t)B_ * T_ * D_;       // 4194304
  const size_t nP = (size_t)P_ * D_;            // 1048064
  const size_t nW = (size_t)D_ * D_;            // 262144
  const size_t nH = (size_t)B_ * H_ * T_ * DK_; // 4194304
  const size_t nPB = (size_t)H_ * P_ * DK_;     // 1048064

  u16* xq  = (u16*)alloc(nX * 2);
  u16* xk  = (u16*)alloc(nX * 2);
  u16* xv  = (u16*)alloc(nX * 2);
  u16* xp  = (u16*)alloc(nP * 2);
  u16* wqb = (u16*)alloc(nW * 2);
  u16* wkb = (u16*)alloc(nW * 2);
  u16* wvb = (u16*)alloc(nW * 2);
  u16* wpb = (u16*)alloc(nW * 2);
  u16* wob = (u16*)alloc(nW * 2);
  u16* qub = (u16*)alloc(nH * 2);
  u16* qvb = (u16*)alloc(nH * 2);
  u16* kbb = (u16*)alloc(nH * 2);
  u16* vtb = (u16*)alloc(nH * 2);
  u16* pbb = (u16*)alloc(nPB * 2);
  u16* ctx = (u16*)alloc(nH * 2);

  // 1) convert inputs + weights to bf16
  auto cvt = [&](const float* s, u16* d, size_t n) {
    cvt_bf16_kernel<<<(unsigned)((n + 255) / 256), 256, 0, stream>>>(s, d, (int)n);
  };
  cvt(query, xq, nX); cvt(key, xk, nX); cvt(value, xv, nX); cvt(pos, xp, nP);
  cvt(Wq, wqb, nW); cvt(Wk, wkb, nW); cvt(Wv, wvb, nW); cvt(Wp, wpb, nW); cvt(Wo, wob, nW);

  // 2) projections (WMMA GEMMs, 32x32 block per wave)
  const int M = B_ * T_;                       // 8192
  const int blocksM = (M / 32) * (D_ / 32) / 4;     // 1024
  const int blocksP = (2048 / 32) * (D_ / 32) / 4;  // 256
  proj_kernel<<<blocksM, 128, 0, stream>>>(xq, wqb, bq, M, M, D_, 0, pbu, pbv,
                                           qub, qvb, nullptr);
  proj_kernel<<<blocksM, 128, 0, stream>>>(xk, wkb, bk, M, M, D_, 1, nullptr, nullptr,
                                           kbb, nullptr, nullptr);
  proj_kernel<<<blocksM, 128, 0, stream>>>(xv, wvb, bv, M, M, D_, 2, nullptr, nullptr,
                                           vtb, nullptr, nullptr);
  proj_kernel<<<blocksP, 128, 0, stream>>>(xp, wpb, nullptr, 2048, P_, D_, 3, nullptr, nullptr,
                                           pbb, nullptr, nullptr);

  // 3) fused relative-position flash attention (4096 waves, 4 per block)
  attn_kernel<<<1024, 128, 0, stream>>>(qub, qvb, kbb, vtb, pbb, mask, ctx);

  // 4) output projection -> f32 d_out
  proj_kernel<<<blocksM, 128, 0, stream>>>(ctx, wob, bo, M, M, D_, 4, nullptr, nullptr,
                                           nullptr, nullptr, out);
}
